// multihead_attention_9148280341045
// MI455X (gfx1250) — matrix operand order repaired and checked by reading, not compiled, not run on hardware
//
#include <hip/hip_runtime.h>
#include <hip/hip_bf16.h>
#include <math.h>

#define HEADS  16
#define DMODEL 1024
#define DH     64
#define BATCH  8
#define SEQ    1024
#define ROWS   (BATCH*SEQ)      /* 8192 */
#define NEG_BIG (-4294967295.0f)

typedef __attribute__((ext_vector_type(16))) _Float16 v16h;
typedef __attribute__((ext_vector_type(8)))  _Float16 v8h;
typedef __attribute__((ext_vector_type(8)))  float    v8f;
typedef __attribute__((vector_size(16)))     int      v4i_t;

// ---- CDNA5 async global->LDS path (guarded; fallback = load + ds_store) ----
#if defined(__has_builtin)
#  if __has_builtin(__builtin_amdgcn_global_load_async_to_lds_b128)
#    define HAVE_ASYNC_LDS 1
#  endif
#endif
#ifndef HAVE_ASYNC_LDS
#  define HAVE_ASYNC_LDS 0
#endif

__device__ __forceinline__ void stage16(const _Float16* gsrc, _Float16* lds) {
#if HAVE_ASYNC_LDS
    __builtin_amdgcn_global_load_async_to_lds_b128(
        (__attribute__((address_space(1))) v4i_t*)gsrc,
        (__attribute__((address_space(3))) v4i_t*)lds, 0, 0);
#else
    *(v8h*)lds = *(const v8h*)gsrc;
#endif
}
#if HAVE_ASYNC_LDS
#  define ASYNC_WAIT(n) asm volatile("s_wait_asynccnt %0" ::"i"(n) : "memory")
#else
#  define ASYNC_WAIT(n)
#endif

// ---------------------------------------------------------------- fragments
__device__ __forceinline__ v16h frag_cat(v8h lo, v8h hi) {
    v16h r;
#pragma unroll
    for (int i = 0; i < 8; ++i) { r[i] = lo[i]; r[8 + i] = hi[i]; }
    return r;
}

// A-fragment 16x32 f16 (MxK), source row-major [M][K], leading dim ld.
// Lane L: M = L&15, H = L>>4; holds K in {8H..8H+7} and {16+8H..16+8H+7}.
__device__ __forceinline__ v16h load_a(const _Float16* base, int ld) {
    int lane = threadIdx.x & 31;
    const _Float16* p = base + (size_t)(lane & 15) * ld + (lane >> 4) * 8;
    return frag_cat(*(const v8h*)p, *(const v8h*)(p + 16));
}

// B-fragment 32x16 f16 (KxN) from "transposed" storage: element (K,N) at
// base[N*ld + K]. Lane L: N = L&15, H = L>>4; holds K in {8H..8H+7} and {16+8H..16+8H+7}.
__device__ __forceinline__ v16h load_bT(const _Float16* base, int ld) {
    int lane = threadIdx.x & 31;
    const _Float16* p = base + (size_t)(lane & 15) * ld + (lane >> 4) * 8;
    return frag_cat(*(const v8h*)p, *(const v8h*)(p + 16));
}

__device__ __forceinline__ v8f wmma16(v16h a, v16h b, v8f c) {
    return __builtin_amdgcn_wmma_f32_16x16x32_f16(false, a, false, b,
                                                  (short)0, c, false, false);
}

// Row-wise reduction across the 16 lanes of a wave32 half (C-fragment N dim).
__device__ __forceinline__ float half_max(float v) {
#pragma unroll
    for (int o = 1; o < 16; o <<= 1) v = fmaxf(v, __shfl_xor(v, o, 32));
    return v;
}
__device__ __forceinline__ float half_sum(float v) {
#pragma unroll
    for (int o = 1; o < 16; o <<= 1) v += __shfl_xor(v, o, 32);
    return v;
}

// ---------------------------------------------------------------- prep
__global__ void __launch_bounds__(256)
prep_rows_kernel(const float* __restrict__ q, const float* __restrict__ k,
                 const float* __restrict__ v, _Float16* __restrict__ qh,
                 _Float16* __restrict__ kh, _Float16* __restrict__ vh,
                 float* __restrict__ qmask, float* __restrict__ kmask) {
    int bid = blockIdx.x;                 // 3*ROWS blocks
    int t = bid / ROWS, row = bid % ROWS;
    const float* src = (t == 0) ? q : (t == 1) ? k : v;
    _Float16* dst = (t == 0) ? qh : (t == 1) ? kh : vh;
    int tid = threadIdx.x;
    float s = 0.f;
#pragma unroll
    for (int j = 0; j < 4; ++j) {
        int i = tid + j * 256;
        float x = src[(size_t)row * DMODEL + i];
        s += x;
        dst[(size_t)row * DMODEL + i] = (_Float16)x;
    }
    if (t == 2) return;                   // block-uniform
#pragma unroll
    for (int o = 1; o < 32; o <<= 1) s += __shfl_xor(s, o, 32);
    __shared__ float red[8];
    if ((tid & 31) == 0) red[tid >> 5] = s;
    __syncthreads();
    if (tid == 0) {
        float tot = 0.f;
#pragma unroll
        for (int i = 0; i < 8; ++i) tot += red[i];
        ((t == 0) ? qmask : kmask)[row] = (tot != 0.f) ? 1.f : 0.f;
    }
}

__global__ void __launch_bounds__(256)
cvt_w_kernel(const float* __restrict__ wq, const float* __restrict__ wk,
             const float* __restrict__ wv, _Float16* __restrict__ hq,
             _Float16* __restrict__ hk, _Float16* __restrict__ hv) {
    size_t base = ((size_t)blockIdx.x * 256 + threadIdx.x) * 4;
    int t = (int)(base >> 20);
    const float* src = (t == 0) ? wq : (t == 1) ? wk : wv;
    _Float16* dst = (t == 0) ? hq : (t == 1) ? hk : hv;
    size_t off = base & ((1u << 20) - 1);
#pragma unroll
    for (int j = 0; j < 4; ++j) dst[off + j] = (_Float16)src[off + j];
}

// ---------------------------------------------------------------- projection
// Y[8192,1024] = relu(X @ W^T + b); async-LDS double-buffered, K-stage = 64.
__global__ void __launch_bounds__(256)
proj_kernel(const _Float16* __restrict__ X, const _Float16* __restrict__ W,
            const float* __restrict__ bias, _Float16* __restrict__ Y) {
    int bid = blockIdx.x;                 // 512 blocks: 64 M-tiles x 8 N-tiles
    int mt = bid >> 3, nt = bid & 7;
    int tid = threadIdx.x;
    int wv = tid >> 5, lane = tid & 31;
    int n = lane & 15, mg = lane >> 4;
    int row0 = mt * 128 + wv * 16;

    __shared__ __align__(16) _Float16 Xs[2][128][64];   // 16 KB each buf
    __shared__ __align__(16) _Float16 Ws[2][128][64];   // 16 KB each buf

    v8f c[8];
#pragma unroll
    for (int f = 0; f < 8; ++f)
#pragma unroll
        for (int r = 0; r < 8; ++r) c[f][r] = 0.f;

    // staging: tile row = 64 f16 = 8 segments of 16B; 1024 segs per tile,
    // 4 per thread per tile (X and W each).
    const int NST = DMODEL / 64;          // 16 stages
    auto stage = [&](int buf, int ks) {
#pragma unroll
        for (int j = 0; j < 4; ++j) {
            int s = tid + j * 256;
            int row = s >> 3, off = (s & 7) * 8;
            stage16(X + (size_t)(mt * 128 + row) * DMODEL + ks * 64 + off,
                    &Xs[buf][row][off]);
        }
#pragma unroll
        for (int j = 0; j < 4; ++j) {
            int s = tid + j * 256;
            int row = s >> 3, off = (s & 7) * 8;
            stage16(W + (size_t)(nt * 128 + row) * DMODEL + ks * 64 + off,
                    &Ws[buf][row][off]);
        }
    };

    stage(0, 0);
    for (int ks = 0; ks < NST; ++ks) {
        int buf = ks & 1;
        if (ks + 1 < NST) { stage(buf ^ 1, ks + 1); ASYNC_WAIT(8); }
        else             { ASYNC_WAIT(0); }
        __syncthreads();
#pragma unroll
        for (int sub = 0; sub < 2; ++sub) {
            v16h a = load_a(&Xs[buf][wv * 16][sub * 32], 64);
#pragma unroll
            for (int f = 0; f < 8; ++f) {
                v16h b = load_bT(&Ws[buf][f * 16][sub * 32], 64);
                c[f] = wmma16(a, b, c[f]);
            }
        }
        __syncthreads();
    }
#pragma unroll
    for (int f = 0; f < 8; ++f) {
        int col = nt * 128 + f * 16 + n;
        float bcol = bias[col];
#pragma unroll
        for (int r = 0; r < 8; ++r) {
            float y = c[f][r] + bcol;
            y = (y > 0.f) ? y : 0.f;
            Y[(size_t)(row0 + r + 8 * mg) * DMODEL + col] = (_Float16)y;
        }
    }
}

// ---------------------------------------------------------------- V transpose
// Vt[b][h][d][s] = V[(b,s)][h*64+d]  (per-head transpose via LDS tile)
__global__ void __launch_bounds__(256)
vtrans_kernel(const _Float16* __restrict__ V, _Float16* __restrict__ Vt) {
    int bid = blockIdx.x;                 // 8*16*16 = 2048
    int b = bid >> 8, h = (bid >> 4) & 15, st = bid & 15;
    __shared__ _Float16 tile[64][72];
    int tid = threadIdx.x;
    int sl = tid >> 2, dc = (tid & 3) * 16;
    const _Float16* src =
        V + (size_t)(b * SEQ + st * 64 + sl) * DMODEL + h * DH + dc;
#pragma unroll
    for (int i = 0; i < 16; ++i) tile[sl][dc + i] = src[i];
    __syncthreads();
    int dl = tid >> 2, sc = (tid & 3) * 16;
    _Float16* dst =
        Vt + (size_t)((b * HEADS + h) * DH + dl) * SEQ + st * 64 + sc;
#pragma unroll
    for (int i = 0; i < 16; ++i) dst[i] = tile[sc + i][dl];
}

// ---------------------------------------------------------------- attention
// Flash-style; K/V tiles async-staged to LDS, shared by all 8 waves.
__global__ void __launch_bounds__(256)
attn_kernel(const _Float16* __restrict__ Q, const _Float16* __restrict__ K,
            const _Float16* __restrict__ Vt, const float* __restrict__ qmask,
            const float* __restrict__ kmask, float* __restrict__ out_pre) {
    int bid = blockIdx.x;                 // 8*16*8 = 1024
    int b = bid >> 7, h = (bid >> 3) & 15, qt = bid & 7;
    int tid = threadIdx.x;
    int wv = tid >> 5, lane = tid & 31;
    int n = lane & 15, mg = lane >> 4;
    int q0 = qt * 128 + wv * 16;

    __shared__ __align__(16) _Float16 Ks[2][32][64];    // 4 KB per buf
    __shared__ __align__(16) _Float16 Vs[2][64][32];    // 4 KB per buf
    __shared__ __align__(16) _Float16 p_lds[8][16][32]; // 8 KB

    // staging: K tile row = 64 f16 = 8 segs; V tile row = 32 f16 = 4 segs.
    auto stage = [&](int buf, int k0) {
        int key = tid >> 3, koff = (tid & 7) * 8;
        stage16(K + (size_t)(b * SEQ + k0 + key) * DMODEL + h * DH + koff,
                &Ks[buf][key][koff]);
        int d = tid >> 2, voff = (tid & 3) * 8;
        stage16(Vt + (size_t)((b * HEADS + h) * DH + d) * SEQ + k0 + voff,
                &Vs[buf][d][voff]);
    };

    v16h aq[2];
#pragma unroll
    for (int d = 0; d < 2; ++d)
        aq[d] = load_a(Q + (size_t)(b * SEQ + q0) * DMODEL + h * DH + d * 32,
                       DMODEL);

    float m_r[8], l_r[8];
    v8f o[4];
#pragma unroll
    for (int r = 0; r < 8; ++r) { m_r[r] = -3.0e38f; l_r[r] = 0.f; }
#pragma unroll
    for (int df = 0; df < 4; ++df)
#pragma unroll
        for (int r = 0; r < 8; ++r) o[df][r] = 0.f;

    stage(0, 0);
    for (int kt = 0; kt < SEQ / 32; ++kt) {
        int buf = kt & 1;
        int k0 = kt * 32;
        if (kt + 1 < SEQ / 32) { stage(buf ^ 1, k0 + 32); ASYNC_WAIT(2); }
        else                   { ASYNC_WAIT(0); }
        __syncthreads();

        // scores: load all 4 K fragments first, then 4 WMMAs
        v16h bk[2][2];
#pragma unroll
        for (int nf = 0; nf < 2; ++nf)
#pragma unroll
            for (int ds = 0; ds < 2; ++ds)
                bk[nf][ds] = load_bT(&Ks[buf][nf * 16][ds * 32], 64);
        v8f s0, s1;
#pragma unroll
        for (int r = 0; r < 8; ++r) { s0[r] = 0.f; s1[r] = 0.f; }
        s0 = wmma16(aq[0], bk[0][0], s0);
        s0 = wmma16(aq[1], bk[0][1], s0);
        s1 = wmma16(aq[0], bk[1][0], s1);
        s1 = wmma16(aq[1], bk[1][1], s1);

        float km0 = kmask[b * SEQ + k0 + n];
        float km1 = kmask[b * SEQ + k0 + 16 + n];
        float alpha[8];
#pragma unroll
        for (int r = 0; r < 8; ++r) {
            float x0 = s0[r] * 0.125f; if (km0 == 0.f) x0 = NEG_BIG;
            float x1 = s1[r] * 0.125f; if (km1 == 0.f) x1 = NEG_BIG;
            float mnew = fmaxf(m_r[r], half_max(fmaxf(x0, x1)));
            float al = __expf(m_r[r] - mnew);
            float p0 = __expf(x0 - mnew), p1 = __expf(x1 - mnew);
            l_r[r] = l_r[r] * al + half_sum(p0 + p1);
            m_r[r] = mnew; alpha[r] = al;
            p_lds[wv][r + 8 * mg][n]      = (_Float16)p0;
            p_lds[wv][r + 8 * mg][16 + n] = (_Float16)p1;
        }
#pragma unroll
        for (int df = 0; df < 4; ++df)
#pragma unroll
            for (int r = 0; r < 8; ++r) o[df][r] *= alpha[r];
        asm volatile("s_wait_dscnt 0x0" ::: "memory");
        v16h pa = load_a(&p_lds[wv][0][0], 32);
#pragma unroll
        for (int df = 0; df < 4; ++df) {
            v16h bv = load_bT(&Vs[buf][df * 16][0], 32);
            o[df] = wmma16(pa, bv, o[df]);
        }
        __syncthreads();
    }
    float sc[8];
#pragma unroll
    for (int r = 0; r < 8; ++r)
        sc[r] = qmask[b * SEQ + q0 + r + 8 * mg] / l_r[r];
#pragma unroll
    for (int df = 0; df < 4; ++df)
#pragma unroll
        for (int r = 0; r < 8; ++r)
            out_pre[(size_t)(b * SEQ + q0 + r + 8 * mg) * DMODEL + h * DH +
                    df * 16 + n] = o[df][r] * sc[r];
}

// ---------------------------------------------------------------- layernorm
__global__ void __launch_bounds__(256)
ln_kernel(const float* __restrict__ pre, const float* __restrict__ qin,
          const float* __restrict__ gamma, const float* __restrict__ beta,
          float* __restrict__ out) {
    __shared__ float rs[8], rss[8];
    __shared__ float mean_s, inv_s;
    int row = blockIdx.x, tid = threadIdx.x;
    float x[4], s = 0.f, ss = 0.f;
#pragma unroll
    for (int j = 0; j < 4; ++j) {
        int i = tid + j * 256;
        float v = pre[(size_t)row * DMODEL + i] + qin[(size_t)row * DMODEL + i];
        x[j] = v; s += v; ss += v * v;
    }
#pragma unroll
    for (int o = 1; o < 32; o <<= 1) {
        s += __shfl_xor(s, o, 32); ss += __shfl_xor(ss, o, 32);
    }
    if ((tid & 31) == 0) { rs[tid >> 5] = s; rss[tid >> 5] = ss; }
    __syncthreads();
    if (tid == 0) {
        float S = 0.f, SS = 0.f;
#pragma unroll
        for (int i = 0; i < 8; ++i) { S += rs[i]; SS += rss[i]; }
        float mean = S / 1024.f;
        float var = (SS - 1024.f * mean * mean) / 1023.f;
        var = (var > 0.f) ? var : 0.f;
        mean_s = mean;
        inv_s = 1.f / (sqrtf(var) + 1e-8f);
    }
    __syncthreads();
    float mean = mean_s, inv = inv_s;
#pragma unroll
    for (int j = 0; j < 4; ++j) {
        int i = tid + j * 256;
        out[(size_t)row * DMODEL + i] = gamma[i] * (x[j] - mean) * inv + beta[i];
    }
}

// ---------------------------------------------------------------- launch
extern "C" void kernel_launch(void* const* d_in, const int* in_sizes, int n_in,
                              void* d_out, int out_size, void* d_ws,
                              size_t ws_size, hipStream_t stream) {
    (void)in_sizes; (void)n_in; (void)out_size; (void)ws_size;
    const float* q  = (const float*)d_in[0];
    const float* k  = (const float*)d_in[1];
    const float* v  = (const float*)d_in[2];
    const float* Wq = (const float*)d_in[3];
    const float* bq = (const float*)d_in[4];
    const float* Wk = (const float*)d_in[5];
    const float* bk = (const float*)d_in[6];
    const float* Wv = (const float*)d_in[7];
    const float* bv = (const float*)d_in[8];
    const float* gamma = (const float*)d_in[9];
    const float* beta  = (const float*)d_in[10];
    float* out = (float*)d_out;

    const size_t XB = (size_t)ROWS * DMODEL * sizeof(_Float16);   // 16 MB
    const size_t WB = (size_t)DMODEL * DMODEL * sizeof(_Float16); //  2 MB
    char* w = (char*)d_ws;
    _Float16* qh  = (_Float16*)(w);                 // reused as Vt later
    _Float16* kh  = (_Float16*)(w + XB);            // kh+vh reused as out_pre
    _Float16* vh  = (_Float16*)(w + 2 * XB);
    _Float16* whq = (_Float16*)(w + 3 * XB);
    _Float16* whk = (_Float16*)(w + 3 * XB + WB);
    _Float16* whv = (_Float16*)(w + 3 * XB + 2 * WB);
    _Float16* Qp  = (_Float16*)(w + 3 * XB + 3 * WB);
    _Float16* Kp  = (_Float16*)(w + 4 * XB + 3 * WB);
    _Float16* Vp  = (_Float16*)(w + 5 * XB + 3 * WB);
    float*    qm  = (float*)(w + 6 * XB + 3 * WB);
    float*    km  = (float*)(w + 6 * XB + 3 * WB + ROWS * sizeof(float));
    _Float16* Vt  = qh;               // safe: qh consumed by proj(Q) first
    float*    pre = (float*)kh;       // safe: kh/vh consumed by proj(K/V)

    prep_rows_kernel<<<3 * ROWS, 256, 0, stream>>>(q, k, v, qh, kh, vh, qm, km);
    cvt_w_kernel<<<3072, 256, 0, stream>>>(Wq, Wk, Wv, whq, whk, whv);
    proj_kernel<<<512, 256, 0, stream>>>(qh, whq, bq, Qp);
    proj_kernel<<<512, 256, 0, stream>>>(kh, whk, bk, Kp);
    proj_kernel<<<512, 256, 0, stream>>>(vh, whv, bv, Vp);
    vtrans_kernel<<<2048, 256, 0, stream>>>(Vp, Vt);
    attn_kernel<<<1024, 256, 0, stream>>>(Qp, Kp, Vt, qm, km, pre);
    ln_kernel<<<ROWS, 256, 0, stream>>>(pre, q, gamma, beta, out);
}
